// S4D_19550691131557
// MI455X (gfx1250) — matrix operand order repaired and checked by reading, not compiled, not run on hardware
//
#include <hip/hip_runtime.h>
#include <hip/hip_bf16.h>
#include <math.h>

typedef __attribute__((ext_vector_type(16))) _Float16 v16h;
typedef __attribute__((ext_vector_type(8)))  float    v8f;

#define B_ 8
#define H_ 512
#define L_ 4096
#define N2_ 32
#define NBLK 4   // N-tiles per wave in the GEMM (A-fragment register reuse)

// ---------------------------------------------------------------------------
// Kernel 1: per-(h,n) discretized parameters.
//   A  = -exp(log_A_real) + i*A_imag
//   dt = exp(log_dt[h]);  dtA = A*dt;  r = exp(dtA)
//   Cdisc = C * (r - 1) / A   (store 2*Cdisc so y = Re(c2 * x))
// ---------------------------------------------------------------------------
__global__ void s4d_params_kernel(const float* __restrict__ C,
                                  const float* __restrict__ log_dt,
                                  const float* __restrict__ log_A_real,
                                  const float* __restrict__ A_imag,
                                  float* __restrict__ rre, float* __restrict__ rim,
                                  float* __restrict__ cre, float* __restrict__ cim) {
    int i = blockIdx.x * blockDim.x + threadIdx.x;
    if (i >= H_ * N2_) return;
    int h = i >> 5;                       // N2_ == 32
    float are = -expf(log_A_real[i]);
    float ai  = A_imag[i];
    float dt  = expf(log_dt[h]);
    float dre = are * dt, dim_ = ai * dt;
    float mag = expf(dre);
    float rr = mag * cosf(dim_);
    float ri = mag * sinf(dim_);
    // E = r - 1;  q = E / A = E * conj(A) / |A|^2
    float Er = rr - 1.0f, Ei = ri;
    float inv = 1.0f / (are * are + ai * ai);
    float qr = (Er * are + Ei * ai) * inv;
    float qi = (Ei * are - Er * ai) * inv;
    float ccr = C[2 * i], cci = C[2 * i + 1];
    float fr = ccr * qr - cci * qi;
    float fi = ccr * qi + cci * qr;
    rre[i] = rr; rim[i] = ri;
    cre[i] = 2.0f * fr; cim[i] = 2.0f * fi;
}

// ---------------------------------------------------------------------------
// Kernel 2: convert W_out (2H x H, f32) to f16 (same layout: K=h contiguous).
// ---------------------------------------------------------------------------
__global__ void s4d_wcvt_kernel(const float* __restrict__ W, _Float16* __restrict__ Wh, int n) {
    int i = blockIdx.x * blockDim.x + threadIdx.x;
    if (i < n) Wh[i] = (_Float16)W[i];
}

// ---------------------------------------------------------------------------
// Kernel 3: diagonal SSM scan. One wave per (b,h) row; lane n owns complex
// state n (N2==32==wave32). Per step: x = r*x + u, cross-lane reduce
// Re(c2*x), + D*u, exact GELU, store f16 TRANSPOSED as y_t[(b*L+l)*H + h]
// so the GEMM B-operand is K(=h)-contiguous.
// ---------------------------------------------------------------------------
__global__ void s4d_scan_kernel(const float* __restrict__ u,
                                const float* __restrict__ rre, const float* __restrict__ rim,
                                const float* __restrict__ cre, const float* __restrict__ cim,
                                const float* __restrict__ D,
                                _Float16* __restrict__ yt) {
    const int lane = threadIdx.x & 31;
    const int wid  = blockIdx.x * (blockDim.x >> 5) + (threadIdx.x >> 5);
    const int b = wid / H_;
    const int h = wid % H_;
    const int pi = h * N2_ + lane;
    const float rr = rre[pi], ri = rim[pi];
    const float c2r = cre[pi], c2i = cim[pi];
    const float d = D[h];
    const float* __restrict__ up = u + ((size_t)b * H_ + h) * L_;
    float xr = 0.0f, xi = 0.0f;
    for (int l0 = 0; l0 < L_; l0 += 32) {
        float uc = up[l0 + lane];          // coalesced chunk of 32 inputs
        float ykeep = 0.0f;
        #pragma unroll 4
        for (int t = 0; t < 32; ++t) {
            float uv = __shfl(uc, t, 32);  // broadcast u[l0+t]
            float nxr = rr * xr - ri * xi + uv;
            float nxi = rr * xi + ri * xr;
            xr = nxr; xi = nxi;
            float c = c2r * xr - c2i * xi; // 2*Re(Cdisc * x) per state
            // butterfly sum over 32 states -> all lanes hold the total
            c += __shfl_xor(c, 16, 32);
            c += __shfl_xor(c,  8, 32);
            c += __shfl_xor(c,  4, 32);
            c += __shfl_xor(c,  2, 32);
            c += __shfl_xor(c,  1, 32);
            if (lane == t) {
                float y = c + d * uv;
                ykeep = 0.5f * y * (1.0f + erff(y * 0.70710678f)); // exact GELU
            }
        }
        // lane holds y at l = l0+lane; transposed f16 store for the GEMM
        yt[((size_t)b * L_ + l0 + lane) * H_ + h] = (_Float16)ykeep;
    }
}

// ---------------------------------------------------------------------------
// Kernel 4: WMMA GEMM  pre[o, n] = sum_h W[o,h]*y[h,n] + b[o], n = b*L+l,
// fused GLU: out[b,h,l] = pre[h,n] * sigmoid(pre[h+H,n]).
// Each wave: one 16-row M tile (a-rows + matching g-rows) x NBLK 16-col
// N tiles. The a/g A-fragments are loaded once per K-step and applied to
// NBLK B-fragments -> 2*NBLK WMMAs per A-load (~21 FLOP/B per lane), and
// W-fragment L2 traffic drops NBLKx.
// ---------------------------------------------------------------------------
union Frag16 { v16h v; uint4 q[2]; };

__global__ void s4d_gemm_glu_kernel(const _Float16* __restrict__ Wh,
                                    const _Float16* __restrict__ yt,
                                    const float* __restrict__ b_out,
                                    float* __restrict__ out) {
    const int lane   = threadIdx.x & 31;
    const int wave   = threadIdx.x >> 5;
    const int ngrp   = blockIdx.x * (blockDim.x >> 5) + wave;   // group of NBLK n-tiles
    const int mtile  = blockIdx.y;                              // 0 .. H/16-1
    const int nl     = lane & 15;
    const int hi     = lane >> 4;                               // K-half select

    // A fragment row: M = lane&15 (both lane halves hold the same row)
    const _Float16* __restrict__ Wa = Wh + (size_t)(mtile * 16 + nl) * H_;
    const _Float16* __restrict__ Wg = Wa + (size_t)H_ * H_;     // g rows: o = m + H
    const _Float16* __restrict__ Bp[NBLK];
    #pragma unroll
    for (int i = 0; i < NBLK; ++i)
        Bp[i] = yt + (size_t)((ngrp * NBLK + i) * 16 + nl) * H_;

    v8f acc_a[NBLK], acc_g[NBLK];
    #pragma unroll
    for (int i = 0; i < NBLK; ++i) { acc_a[i] = (v8f){}; acc_g[i] = (v8f){}; }

    for (int kk = 0; kk < H_; kk += 32) {
        Frag16 a, g;
        // A 16x32 f16 layout: VGPR j holds K = (j<4 ? 2j : 8+2j) + hi*8
        // -> two contiguous 8-half (16B) groups at kk+hi*8 and kk+16+hi*8
        a.q[0] = *(const uint4*)(Wa + kk + hi * 8);
        a.q[1] = *(const uint4*)(Wa + kk + 16 + hi * 8);
        g.q[0] = *(const uint4*)(Wg + kk + hi * 8);
        g.q[1] = *(const uint4*)(Wg + kk + 16 + hi * 8);
        Frag16 bf[NBLK];
        #pragma unroll
        for (int i = 0; i < NBLK; ++i) {
            // B 32x16 f16 layout: same K order as A -> 8-half groups at kk+hi*8 and kk+16+hi*8
            bf[i].q[0] = *(const uint4*)(Bp[i] + kk + hi * 8);
            bf[i].q[1] = *(const uint4*)(Bp[i] + kk + 16 + hi * 8);
        }
        #pragma unroll
        for (int i = 0; i < NBLK; ++i) {
            acc_a[i] = __builtin_amdgcn_wmma_f32_16x16x32_f16(
                false, a.v, false, bf[i].v, (short)0, acc_a[i], false, false);
            acc_g[i] = __builtin_amdgcn_wmma_f32_16x16x32_f16(
                false, g.v, false, bf[i].v, (short)0, acc_g[i], false, false);
        }
    }

    // Epilogue: bias + GLU, scatter to (B,H,L) f32 output.
    #pragma unroll
    for (int i = 0; i < NBLK; ++i) {
        const int n  = (ngrp * NBLK + i) * 16 + nl;
        const int bb = n >> 12;          // n / L_
        const int ll = n & (L_ - 1);     // n % L_
        #pragma unroll
        for (int r = 0; r < 8; ++r) {
            int m = mtile * 16 + r + hi * 8;        // D tile: M = r + 8*(lane>=16)
            float av = acc_a[i][r] + b_out[m];
            float gv = acc_g[i][r] + b_out[m + H_];
            float o  = av * (1.0f / (1.0f + expf(-gv)));
            out[((size_t)bb * H_ + m) * L_ + ll] = o;
        }
    }
}

// ---------------------------------------------------------------------------
extern "C" void kernel_launch(void* const* d_in, const int* in_sizes, int n_in,
                              void* d_out, int out_size, void* d_ws, size_t ws_size,
                              hipStream_t stream) {
    const float* u        = (const float*)d_in[0];
    const float* C        = (const float*)d_in[1];
    const float* log_dt   = (const float*)d_in[2];
    const float* log_A    = (const float*)d_in[3];
    const float* A_imag   = (const float*)d_in[4];
    const float* D        = (const float*)d_in[5];
    const float* W_out    = (const float*)d_in[6];
    const float* b_out    = (const float*)d_in[7];
    float* out            = (float*)d_out;

    // workspace layout (bytes)
    char* ws = (char*)d_ws;
    float*    rre = (float*)(ws);                  // 16384 f32
    float*    rim = (float*)(ws + 65536);
    float*    cre = (float*)(ws + 131072);
    float*    cim = (float*)(ws + 196608);
    _Float16* Wh  = (_Float16*)(ws + 262144);      // 2H*H f16 = 1 MB
    _Float16* yt  = (_Float16*)(ws + 1310720);     // (B*L, H) f16 = 32 MB

    s4d_params_kernel<<<(H_ * N2_ + 255) / 256, 256, 0, stream>>>(
        C, log_dt, log_A, A_imag, rre, rim, cre, cim);

    s4d_wcvt_kernel<<<(2 * H_ * H_ + 255) / 256, 256, 0, stream>>>(
        W_out, Wh, 2 * H_ * H_);

    // one wave per (b,h): 4096 waves, 8 waves per block
    s4d_scan_kernel<<<(B_ * H_) / 8, 256, 0, stream>>>(
        u, rre, rim, cre, cim, D, yt);

    // GEMM: n-tile groups = (B*L/16)/NBLK = 512 (4 waves/block), mtiles = 32
    dim3 grid(((B_ * L_ / 16) / NBLK) / 4, H_ / 16);
    s4d_gemm_glu_kernel<<<grid, 128, 0, stream>>>(Wh, yt, b_out, out);
}
